// OfficialTTTLayer_62826781606583
// MI455X (gfx1250) — matrix operand order repaired and checked by reading, not compiled, not run on hardware
//
#include <hip/hip_runtime.h>
#include <hip/hip_bf16.h>
#include <cstdint>

typedef __attribute__((ext_vector_type(16))) _Float16 v16h;
typedef __attribute__((ext_vector_type(8)))  _Float16 v8h;
typedef __attribute__((ext_vector_type(8)))  float    v8f;
typedef __attribute__((ext_vector_type(2)))  float    v2f;

#define HIDN  1024
#define SEQ   2048
#define BATCH 4
#define NH    16
#define HD    64
#define MBS   16
#define NMB   128

// ------------------------- CDNA5 async-copy helpers -------------------------
// GLOBAL_LOAD_ASYNC_TO_LDS_B128: memory -> LDS directly, tracked by ASYNCcnt.
__device__ __forceinline__ void async_b128(uint32_t lds_byte_addr, const void* gaddr) {
    asm volatile("global_load_async_to_lds_b128 %0, %1, off"
                 :: "v"(lds_byte_addr), "v"((uint64_t)(uintptr_t)gaddr)
                 : "memory");
}
__device__ __forceinline__ void wait_async0() {
    asm volatile("s_wait_asynccnt 0x0" ::: "memory");
}
__device__ __forceinline__ void wait_async8() {
    asm volatile("s_wait_asynccnt 0x8" ::: "memory");
}
__device__ __forceinline__ uint32_t lds_addr(const void* p) {
    return (uint32_t)(uintptr_t)p;   // low 32 bits of flat LDS pointer = LDS offset
}

// ---------------------------------------------------------------------------
// f32 -> f16 conversion pass (bandwidth trivial; removes cvt from GEMM loop)
// ---------------------------------------------------------------------------
__global__ __launch_bounds__(256) void f32_to_f16_kernel(
    const float* __restrict__ src, _Float16* __restrict__ dst, int n4)
{
    int i = blockIdx.x * 256 + threadIdx.x;
    if (i < n4) {
        float4 v = ((const float4*)src)[i];
        union { _Float16 h[4]; uint2 u; } t;
        t.h[0] = (_Float16)v.x; t.h[1] = (_Float16)v.y;
        t.h[2] = (_Float16)v.z; t.h[3] = (_Float16)v.w;
        ((uint2*)dst)[i] = t.u;
    }
}

// ---------------------------------------------------------------------------
// C = A @ W^T, A: MxK f16, W: NxK f16, C: MxN f32.
// 128x128x64 tiles, 8 wave32, double-buffered async-to-LDS staging.
// blockIdx.z indexes weight z (W + z*N*K) and output z (D + z*M*N).
// ---------------------------------------------------------------------------
#define BM 128
#define BN 128
#define BK 64
#define LDT 72   // padded LDS row stride in halves (144B -> spread banks)

__global__ __launch_bounds__(256) void wmma_gemm_h(
    const _Float16* __restrict__ Ah,
    const _Float16* __restrict__ Whb,
    float* __restrict__ Db,
    int M, int N, int K)
{
    const _Float16* Wh = Whb + (size_t)blockIdx.z * N * K;
    float*          D  = Db  + (size_t)blockIdx.z * M * N;

    __shared__ __align__(16) _Float16 As[2][BM * LDT];
    __shared__ __align__(16) _Float16 Bs[2][BN * LDT];

    const int tid   = threadIdx.x;
    const int wave  = tid >> 5;
    const int lane  = tid & 31;
    const int lhalf = lane >> 4;
    const int lidx  = lane & 15;
    const int bm = blockIdx.x * BM;
    const int bn = blockIdx.y * BN;
    const int wm = (wave >> 1) * 32;
    const int wn = (wave & 1) * 64;

    v8f acc[2][4];
    {
        v8f z = {};
        #pragma unroll
        for (int i = 0; i < 2; ++i)
            #pragma unroll
            for (int j = 0; j < 4; ++j) acc[i][j] = z;
    }

    // copy mapping: 2 threads per row, 32 halves each, 4x b128 per matrix
    const int lrow = tid >> 1;
    const int lch  = (tid & 1) * 32;
    const _Float16* gA = Ah + (size_t)(bm + lrow) * K + lch;
    const _Float16* gB = Wh + (size_t)(bn + lrow) * K + lch;
    const uint32_t sAo[2] = { lds_addr(&As[0][lrow * LDT + lch]),
                              lds_addr(&As[1][lrow * LDT + lch]) };
    const uint32_t sBo[2] = { lds_addr(&Bs[0][lrow * LDT + lch]),
                              lds_addr(&Bs[1][lrow * LDT + lch]) };

    auto issue_tile = [&](int k0, int buf) {
        #pragma unroll
        for (int i = 0; i < 4; ++i) {
            async_b128(sAo[buf] + 16 * i, gA + k0 + 8 * i);
            async_b128(sBo[buf] + 16 * i, gB + k0 + 8 * i);
        }
    };

    const int ktiles = K / BK;
    issue_tile(0, 0);
    for (int kt = 0; kt < ktiles; ++kt) {
        const int cur = kt & 1;
        if (kt + 1 < ktiles) {           // prefetch next tile into other buffer
            issue_tile((kt + 1) * BK, cur ^ 1);
            wait_async8();               // previous 8 (this tile) complete, in order
        } else {
            wait_async0();
        }
        __syncthreads();

        const _Float16* Ab = As[cur];
        const _Float16* Bb = Bs[cur];
        #pragma unroll
        for (int s = 0; s < 2; ++s) {
            union { v16h v; v8h h[2]; } af[2], bf[4];
            #pragma unroll
            for (int ti = 0; ti < 2; ++ti) {
                const _Float16* p = Ab + (wm + 16 * ti + lidx) * LDT + s * 32 + lhalf * 8;
                af[ti].h[0] = *(const v8h*)p;
                af[ti].h[1] = *(const v8h*)(p + 16);
            }
            #pragma unroll
            for (int tj = 0; tj < 4; ++tj) {
                const _Float16* p = Bb + (wn + 16 * tj + lidx) * LDT + s * 32 + lhalf * 8;
                bf[tj].h[0] = *(const v8h*)p;
                bf[tj].h[1] = *(const v8h*)(p + 16);
            }
            #pragma unroll
            for (int ti = 0; ti < 2; ++ti)
                #pragma unroll
                for (int tj = 0; tj < 4; ++tj)
                    acc[ti][tj] = __builtin_amdgcn_wmma_f32_16x16x32_f16(
                        false, af[ti].v, false, bf[tj].v,
                        (short)0, acc[ti][tj], false, false);
        }
        __syncthreads();
    }

    #pragma unroll
    for (int ti = 0; ti < 2; ++ti)
        #pragma unroll
        for (int tj = 0; tj < 4; ++tj) {
            int row0 = bm + wm + 16 * ti + 8 * lhalf;
            int col  = bn + wn + 16 * tj + lidx;
            #pragma unroll
            for (int r = 0; r < 8; ++r)
                D[(size_t)(row0 + r) * N + col] = acc[ti][tj][r];
        }
}

// ---------------------------------------------------------------------------
// TTT scan: one block per (b,h), 4 wave32, fp32 WMMA 16x16x4.
// q/k/v chunks double-buffered in LDS; chunk c+1 prefetched with async-to-LDS
// while the serial chunk-c recurrence computes (hides global latency behind
// the 128-step chain).
// ---------------------------------------------------------------------------
__device__ __forceinline__ v8f wmma_f32x4(v2f a, v2f b, v8f c) {
    return __builtin_amdgcn_wmma_f32_16x16x4_f32(false, a, false, b,
                                                 (short)0, c, false, false);
}

__global__ __launch_bounds__(128) void ttt_scan(
    const float* __restrict__ Q,
    const float* __restrict__ Kp,
    const float* __restrict__ V,
    float* __restrict__ Yout,          // aliases Q (in-place)
    const float* __restrict__ W1i,     // (H,D,D)
    const float* __restrict__ b1i,     // (H,1,D)
    const float* __restrict__ tg,      // (D)
    const float* __restrict__ tb)      // (D)
{
    __shared__ __align__(16) float W1s[HD][HD];
    __shared__ float b1s[HD], tgs[HD], tbs[HD];
    __shared__ __align__(16) float qs[2][MBS][HD], ks[2][MBS][HD], vs[2][MBS][HD];
    __shared__ __align__(16) float zs[MBS][HD], gs[MBS][HD];
    __shared__ float comb[MBS][MBS];
    __shared__ float mu[MBS], rstd[MBS];

    const int tid   = threadIdx.x;
    const int wave  = tid >> 5;
    const int lane  = tid & 31;
    const int lhalf = lane >> 4;
    const int lidx  = lane & 15;
    const int nw    = wave * 16;

    const int b = blockIdx.x >> 4;
    const int h = blockIdx.x & 15;

    // init fast weights / constants
    {
        const float4* w1p = (const float4*)(W1i + (size_t)h * HD * HD);
        for (int i = tid; i < HD * HD / 4; i += 128)
            ((float4*)&W1s[0][0])[i] = w1p[i];
        if (tid < HD) {
            b1s[tid] = b1i[h * HD + tid];
            tgs[tid] = tg[tid];
            tbs[tid] = tb[tid];
        }
    }

    const size_t base = (size_t)b * SEQ * HIDN + (size_t)h * HD;
    // per-thread chunk-copy mapping: 2 float4 slots per matrix
    const int i0 = tid * 2, i1 = tid * 2 + 1;
    const int m0 = i0 >> 4, p0 = (i0 & 15) * 4;
    const int m1 = i1 >> 4, p1 = (i1 & 15) * 4;

    auto issue_chunk = [&](int c, int buf) {
        const size_t cb = base + (size_t)c * MBS * HIDN;
        const size_t o0 = cb + (size_t)m0 * HIDN + p0;
        const size_t o1 = cb + (size_t)m1 * HIDN + p1;
        async_b128(lds_addr(&qs[buf][m0][p0]), Q  + o0);
        async_b128(lds_addr(&qs[buf][m1][p1]), Q  + o1);
        async_b128(lds_addr(&ks[buf][m0][p0]), Kp + o0);
        async_b128(lds_addr(&ks[buf][m1][p1]), Kp + o1);
        async_b128(lds_addr(&vs[buf][m0][p0]), V  + o0);
        async_b128(lds_addr(&vs[buf][m1][p1]), V  + o1);
    };

    issue_chunk(0, 0);
    wait_async0();
    __syncthreads();

    for (int c = 0; c < NMB; ++c) {
        const int cur = c & 1;
        if (c + 1 < NMB) issue_chunk(c + 1, cur ^ 1);   // overlap with compute

        const float (*q)[HD] = qs[cur];
        const float (*k)[HD] = ks[cur];
        const float (*v)[HD] = vs[cur];
        const size_t cbase = base + (size_t)c * MBS * HIDN;

        // Z1 = k @ W1 + b1
        v8f accz = {};
        #pragma unroll
        for (int kk = 0; kk < 16; ++kk) {
            int kA = 4 * kk + 2 * lhalf;
            v2f a, bb;
            a.x  = k[lidx][kA];           a.y  = k[lidx][kA + 1];
            bb.x = W1s[kA][nw + lidx];    bb.y = W1s[kA + 1][nw + lidx];
            accz = wmma_f32x4(a, bb, accz);
        }
        #pragma unroll
        for (int r = 0; r < 8; ++r)
            zs[8 * lhalf + r][nw + lidx] = accz[r] + b1s[nw + lidx];
        __syncthreads();

        // LN stats of Z1 rows
        if (tid < MBS) {
            float s = 0.f;
            for (int d = 0; d < HD; ++d) s += zs[tid][d];
            float m_ = s * (1.0f / HD), v_ = 0.f;
            for (int d = 0; d < HD; ++d) { float t = zs[tid][d] - m_; v_ += t * t; }
            mu[tid]   = m_;
            rstd[tid] = rsqrtf(v_ * (1.0f / HD) + 1e-5f);
        }
        __syncthreads();

        // grad = LN(Z1) - (v - k)
        {
            int m = tid >> 3, d0 = (tid & 7) * 8;
            #pragma unroll
            for (int i = 0; i < 8; ++i) {
                int d = d0 + i;
                float ln = (zs[m][d] - mu[m]) * rstd[m] * tgs[d] + tbs[d];
                gs[m][d] = ln - (v[m][d] - k[m][d]);
            }
        }
        __syncthreads();

        // wave 0: Attn = q @ k^T ; comb[m][j] = -eta_m^2*(Attn+1) on tril
        if (wave == 0) {
            v8f acca = {};
            #pragma unroll
            for (int kk = 0; kk < 16; ++kk) {
                int kA = 4 * kk + 2 * lhalf;
                v2f a, bb;
                a.x  = q[lidx][kA];  a.y  = q[lidx][kA + 1];
                bb.x = k[lidx][kA];  bb.y = k[lidx][kA + 1];
                acca = wmma_f32x4(a, bb, acca);
            }
            #pragma unroll
            for (int r = 0; r < 8; ++r) {
                int m = 8 * lhalf + r, j = lidx;
                float e = 0.1f / ((float)(m + 1) * 64.0f);
                comb[m][j] = (j <= m) ? (-(e * e) * (acca[r] + 1.0f)) : 0.0f;
            }
        }
        // all waves: q @ W1 (old W1)
        v8f accb = {};
        #pragma unroll
        for (int kk = 0; kk < 16; ++kk) {
            int kA = 4 * kk + 2 * lhalf;
            v2f a, bb;
            a.x  = q[lidx][kA];           a.y  = q[lidx][kA + 1];
            bb.x = W1s[kA][nw + lidx];    bb.y = W1s[kA + 1][nw + lidx];
            accb = wmma_f32x4(a, bb, accb);
        }
        __syncthreads();   // comb ready; all W1 reads complete

        // Z1_bar = q@W1 + comb@grad + b1
        #pragma unroll
        for (int kk = 0; kk < 4; ++kk) {
            int kA = 4 * kk + 2 * lhalf;
            v2f a, bb;
            a.x  = comb[lidx][kA];     a.y  = comb[lidx][kA + 1];
            bb.x = gs[kA][nw + lidx];  bb.y = gs[kA + 1][nw + lidx];
            accb = wmma_f32x4(a, bb, accb);
        }
        #pragma unroll
        for (int r = 0; r < 8; ++r)
            zs[8 * lhalf + r][nw + lidx] = accb[r] + b1s[nw + lidx];
        __syncthreads();

        if (tid < MBS) {
            float s = 0.f;
            for (int d = 0; d < HD; ++d) s += zs[tid][d];
            float m_ = s * (1.0f / HD), v_ = 0.f;
            for (int d = 0; d < HD; ++d) { float t = zs[tid][d] - m_; v_ += t * t; }
            mu[tid]   = m_;
            rstd[tid] = rsqrtf(v_ * (1.0f / HD) + 1e-5f);
        }
        __syncthreads();

        // out = q + LN(Z1_bar)
        {
            int m = tid >> 3, d0 = (tid & 7) * 8;
            size_t rb = cbase + (size_t)m * HIDN;
            #pragma unroll
            for (int i = 0; i < 8; ++i) {
                int d = d0 + i;
                float ln = (zs[m][d] - mu[m]) * rstd[m] * tgs[d] + tbs[d];
                Yout[rb + d] = q[m][d] + ln;
            }
        }

        // W1 -= last_eta * k^T @ grad
        {
            const float last_eta = 0.1f / (16.0f * 64.0f);
            #pragma unroll
            for (int mt = 0; mt < 4; ++mt) {
                v8f accw = {};
                #pragma unroll
                for (int kk = 0; kk < 4; ++kk) {
                    int kA = 4 * kk + 2 * lhalf;
                    v2f a, bb;
                    a.x  = k[kA][16 * mt + lidx];  a.y  = k[kA + 1][16 * mt + lidx];
                    bb.x = gs[kA][nw + lidx];      bb.y = gs[kA + 1][nw + lidx];
                    accw = wmma_f32x4(a, bb, accw);
                }
                #pragma unroll
                for (int r = 0; r < 8; ++r)
                    W1s[16 * mt + 8 * lhalf + r][nw + lidx] -= last_eta * accw[r];
            }
        }
        // b1 -= sum_m eta_m * grad[m]
        if (tid < HD) {
            float s = 0.f;
            #pragma unroll
            for (int m = 0; m < MBS; ++m)
                s += (0.1f / ((float)(m + 1) * 64.0f)) * gs[m][tid];
            b1s[tid] -= s;
        }
        if (c + 1 < NMB) wait_async0();   // next chunk resident before flip
        __syncthreads();
    }
}

// ---------------------------------------------------------------------------
// Fused final LN + tanh-GELU gate; writes f16 directly for the Wo GEMM.
// ---------------------------------------------------------------------------
__global__ __launch_bounds__(256) void gated_ln_gelu(
    const float* __restrict__ Y, const float* __restrict__ G,
    const float* __restrict__ lg, const float* __restrict__ lb,
    _Float16* __restrict__ Zh)
{
    const int row = blockIdx.x;
    const int tid = threadIdx.x;
    const float* y = Y + (size_t)row * HIDN;
    const float* g = G + (size_t)row * HIDN;
    __shared__ float red[256];

    float v[4]; float s = 0.f;
    #pragma unroll
    for (int i = 0; i < 4; ++i) { v[i] = y[tid * 4 + i]; s += v[i]; }
    red[tid] = s; __syncthreads();
    for (int st = 128; st > 0; st >>= 1) {
        if (tid < st) red[tid] += red[tid + st];
        __syncthreads();
    }
    float mean = red[0] * (1.0f / HIDN);
    __syncthreads();
    float s2 = 0.f;
    #pragma unroll
    for (int i = 0; i < 4; ++i) { float t = v[i] - mean; s2 += t * t; }
    red[tid] = s2; __syncthreads();
    for (int st = 128; st > 0; st >>= 1) {
        if (tid < st) red[tid] += red[tid + st];
        __syncthreads();
    }
    float rst = rsqrtf(red[0] * (1.0f / HIDN) + 1e-5f);

    union { _Float16 h[4]; uint2 u; } t;
    #pragma unroll
    for (int i = 0; i < 4; ++i) {
        int d = tid * 4 + i;
        float ln = (v[i] - mean) * rst * lg[d] + lb[d];
        float gv = g[d];
        float gl = 0.5f * gv * (1.0f + tanhf(0.7978845608028654f *
                                             (gv + 0.044715f * gv * gv * gv)));
        t.h[i] = (_Float16)(gl * ln);
    }
    ((uint2*)(Zh + (size_t)row * HIDN))[tid] = t.u;
}

// ---------------------------------------------------------------------------
extern "C" void kernel_launch(void* const* d_in, const int* in_sizes, int n_in,
                              void* d_out, int out_size, void* d_ws, size_t ws_size,
                              hipStream_t stream)
{
    const float* x   = (const float*)d_in[0];
    const float* Wq  = (const float*)d_in[1];
    const float* Wk  = (const float*)d_in[2];
    const float* Wv  = (const float*)d_in[3];
    const float* Wg  = (const float*)d_in[4];
    const float* Wo  = (const float*)d_in[5];
    const float* W1i = (const float*)d_in[6];
    const float* b1i = (const float*)d_in[7];
    const float* tg  = (const float*)d_in[8];
    const float* tb  = (const float*)d_in[9];
    const float* lg  = (const float*)d_in[10];
    const float* lb  = (const float*)d_in[11];
    float* out = (float*)d_out;

    const size_t tokens = (size_t)BATCH * SEQ;   // 8192
    const size_t mat    = tokens * HIDN;         // 8M elements
    const size_t wmat   = (size_t)HIDN * HIDN;   // 1M elements

    float* Qb = (float*)d_ws;          // fp32, also Y (in-place)
    float* Kb = Qb + mat;
    float* Vb = Kb + mat;
    float* Gb = Vb + mat;
    _Float16* Xh = (_Float16*)(Gb + mat);         // f16 x
    _Float16* Wh = Xh + mat;                      // f16 Wq,Wk,Wv,Wg,Wo contiguous
    _Float16* Zh = (_Float16*)Kb;                 // reuse K (dead after scan)

    // f16 conversions (one-shot, bandwidth-trivial)
    {
        int n4 = (int)(mat / 4);
        f32_to_f16_kernel<<<dim3((n4 + 255) / 256), 256, 0, stream>>>(x, Xh, n4);
        int w4 = (int)(wmat / 4);
        dim3 wg((w4 + 255) / 256);
        f32_to_f16_kernel<<<wg, 256, 0, stream>>>(Wq, Wh + 0 * wmat, w4);
        f32_to_f16_kernel<<<wg, 256, 0, stream>>>(Wk, Wh + 1 * wmat, w4);
        f32_to_f16_kernel<<<wg, 256, 0, stream>>>(Wv, Wh + 2 * wmat, w4);
        f32_to_f16_kernel<<<wg, 256, 0, stream>>>(Wg, Wh + 3 * wmat, w4);
        f32_to_f16_kernel<<<wg, 256, 0, stream>>>(Wo, Wh + 4 * wmat, w4);
    }

    // Q,K,V,Gate projections (z selects weight & destination; Qb..Gb contiguous)
    dim3 g1((unsigned)(tokens / BM), HIDN / BN, 4);
    wmma_gemm_h<<<g1, 256, 0, stream>>>(Xh, Wh, Qb, (int)tokens, HIDN, HIDN);

    ttt_scan<<<dim3(BATCH * NH), 128, 0, stream>>>(Qb, Kb, Vb, Qb,
                                                   W1i, b1i, tg, tb);

    gated_ln_gelu<<<dim3((unsigned)tokens), 256, 0, stream>>>(Qb, Gb, lg, lb, Zh);

    dim3 g2((unsigned)(tokens / BM), HIDN / BN, 1);
    wmma_gemm_h<<<g2, 256, 0, stream>>>(Zh, Wh + 4 * wmat, out,
                                        (int)tokens, HIDN, HIDN);
}
